// CrossViewContrast_69896297775438
// MI455X (gfx1250) — matrix operand order repaired and checked by reading, not compiled, not run on hardware
//
#include <hip/hip_runtime.h>

typedef __attribute__((ext_vector_type(16))) _Float16 v16h;
typedef __attribute__((ext_vector_type(8)))  _Float16 v8h;
typedef __attribute__((ext_vector_type(4)))  _Float16 v4h;
typedef __attribute__((ext_vector_type(8)))  float    v8f;
typedef __attribute__((ext_vector_type(4)))  int      v4i;

#define NROWS 16384
#define DIM   128
#define ROWS_PER_BLOCK 256
#define NSPLIT 8
#define COLS_PER_SPLIT (NROWS / NSPLIT)
#define STAGE_COLS 64
#define STAGE_STRIDE 272   /* 256B row + 16B pad: conflict-free LDS banks */
#define STAGE_BYTES (STAGE_COLS * STAGE_STRIDE)
#define NSTAGES (COLS_PER_SPLIT / STAGE_COLS)
// exp(x / tau) = exp2(x * (1/tau) / ln2), tau = 0.5. Folded into packed A.
#define EXP2_SCALE 2.8853900817779268f
#define TAU_INV    2.0f

#define HAS_ASYNC_LDS __has_builtin(__builtin_amdgcn_global_load_async_to_lds_b128)

#if HAS_ASYNC_LDS
// Exact parameter types per hipcc diagnostic: v4i in addrspace(1) / addrspace(3).
typedef __attribute__((address_space(1))) v4i as1_v4i;
typedef __attribute__((address_space(3))) v4i as3_v4i;
#endif

__device__ __forceinline__ void wait_async0() {
#if __has_builtin(__builtin_amdgcn_s_wait_asynccnt)
  __builtin_amdgcn_s_wait_asynccnt(0);
#else
  asm volatile("s_wait_asynccnt 0" ::: "memory");
#endif
}

// ---------------------------------------------------------------------------
// Kernel 1: L2-normalize rows of z1,z2 -> f16 packed operands. A rows are
// pre-scaled by EXP2_SCALE so the GEMM epilogue is a bare exp2. Exact f32
// diagonal dot (positives); zero the rowsum accumulator.
// ---------------------------------------------------------------------------
__global__ void __launch_bounds__(256)
norm_pack_kernel(const float* __restrict__ z1, const float* __restrict__ z2,
                 _Float16* __restrict__ Ah, _Float16* __restrict__ Bh,
                 float* __restrict__ posdot, float* __restrict__ rowsum) {
  const int w    = threadIdx.x >> 5;
  const int lane = threadIdx.x & 31;
  const int row  = blockIdx.x * 8 + w;

  const float4 x = ((const float4*)(z1 + (size_t)row * DIM))[lane];
  const float4 y = ((const float4*)(z2 + (size_t)row * DIM))[lane];

  float ss1 = x.x * x.x + x.y * x.y + x.z * x.z + x.w * x.w;
  float ss2 = y.x * y.x + y.y * y.y + y.z * y.z + y.w * y.w;
  float cr  = x.x * y.x + x.y * y.y + x.z * y.z + x.w * y.w;
  #pragma unroll
  for (int m = 1; m < 32; m <<= 1) {
    ss1 += __shfl_xor(ss1, m, 32);
    ss2 += __shfl_xor(ss2, m, 32);
    cr  += __shfl_xor(cr,  m, 32);
  }
  const float s1u = 1.0f / fmaxf(sqrtf(ss1), 1e-12f);
  const float s2  = 1.0f / fmaxf(sqrtf(ss2), 1e-12f);
  const float s1  = s1u * EXP2_SCALE;  // fold exp2 scale into packed A only

  v4h ha, hb;
  ha[0] = (_Float16)(x.x * s1); ha[1] = (_Float16)(x.y * s1);
  ha[2] = (_Float16)(x.z * s1); ha[3] = (_Float16)(x.w * s1);
  hb[0] = (_Float16)(y.x * s2); hb[1] = (_Float16)(y.y * s2);
  hb[2] = (_Float16)(y.z * s2); hb[3] = (_Float16)(y.w * s2);
  ((v4h*)(Ah + (size_t)row * DIM))[lane] = ha;
  ((v4h*)(Bh + (size_t)row * DIM))[lane] = hb;

  if (lane == 0) {
    posdot[row] = cr * s1u * s2;   // exact f32 diagonal: log(pos) = dot / tau
    rowsum[row] = 0.0f;
  }
}

// ---------------------------------------------------------------------------
// Kernel 2: fused sim-GEMM + exp + row-sum, LDS double-buffered.
// Block = 8 waves; wave owns 32 rows (two A fragment sets register-resident
// for the whole sweep -> each B fragment feeds 2 WMMAs). B is staged through
// LDS in 64-column double-buffered stages filled with async-to-LDS copies
// (ASYNCcnt) overlapped with WMMA+exp compute; consumers use short-latency
// ds_loads. One barrier per stage.
//
// Fragment layouts per CDNA5 ISA 7.12.2 (wave32):
//   A 16x32 f16:  lane(r,hi): elems 0..7  = A[r][32k + 8hi + 0..7]
//                              elems 8..15 = A[r][32k + 16 + 8hi + 0..7]
//   B 32x16 f16:  lane(r,hi): elems 0..7 = Bcol[r][32k + 8hi + 0..7], 8..15 = Bcol[r][32k + 16 + 8hi + 0..7]
//   C/D 16x16 f32: lane(r,hi): vgpr v = D[M = v + 8hi][N = r]
// ---------------------------------------------------------------------------
__global__ void __launch_bounds__(256)
simsum_kernel(const _Float16* __restrict__ Ah, const _Float16* __restrict__ Bh,
              float* __restrict__ rowsum) {
  __shared__ alignas(16) char smem[2 * STAGE_BYTES];

  const int tid  = threadIdx.x;
  const int w    = tid >> 5;
  const int lane = tid & 31;
  const int r    = lane & 15;
  const int hi   = lane >> 4;
  const int rowA = blockIdx.x * ROWS_PER_BLOCK + w * 32;

  // Fill-slice mapping: thread copies 64 contiguous bytes of one B row.
  const int frow = tid >> 2;
  const int foff = (tid & 3) * 64;
  const char* Bbytes = (const char*)Bh;
  const int colSplitBase = blockIdx.y * COLS_PER_SPLIT;

  // Two A fragment sets: rows [rowA, rowA+16) and [rowA+16, rowA+32).
  v16h a0[4], a1[4];
  {
    const _Float16* ap0 = Ah + (size_t)(rowA + r) * DIM + 8 * hi;
    const _Float16* ap1 = ap0 + (size_t)16 * DIM;
    #pragma unroll
    for (int k = 0; k < 4; ++k) {
      v8h lo0 = *(const v8h*)(ap0 + 32 * k);
      v8h hi0 = *(const v8h*)(ap0 + 32 * k + 16);
      a0[k] = __builtin_shufflevector(lo0, hi0, 0, 1, 2, 3, 4, 5, 6, 7,
                                                8, 9, 10, 11, 12, 13, 14, 15);
      v8h lo1 = *(const v8h*)(ap1 + 32 * k);
      v8h hi1 = *(const v8h*)(ap1 + 32 * k + 16);
      a1[k] = __builtin_shufflevector(lo1, hi1, 0, 1, 2, 3, 4, 5, 6, 7,
                                                8, 9, 10, 11, 12, 13, 14, 15);
    }
  }

  float racc0[8] = {0.f, 0.f, 0.f, 0.f, 0.f, 0.f, 0.f, 0.f};
  float racc1[8] = {0.f, 0.f, 0.f, 0.f, 0.f, 0.f, 0.f, 0.f};

  // --- Prologue: fill stage 0 ---
  {
    const char* g = Bbytes + (size_t)(colSplitBase + frow) * 256 + foff;
    char* l = smem + frow * STAGE_STRIDE + foff;
#if HAS_ASYNC_LDS
    __builtin_amdgcn_global_load_async_to_lds_b128((as1_v4i*)g, (as3_v4i*)l, 0, 0);
    __builtin_amdgcn_global_load_async_to_lds_b128((as1_v4i*)g, (as3_v4i*)l, 16, 0);
    __builtin_amdgcn_global_load_async_to_lds_b128((as1_v4i*)g, (as3_v4i*)l, 32, 0);
    __builtin_amdgcn_global_load_async_to_lds_b128((as1_v4i*)g, (as3_v4i*)l, 48, 0);
    wait_async0();
#else
    ((uint4*)l)[0] = ((const uint4*)g)[0];
    ((uint4*)l)[1] = ((const uint4*)g)[1];
    ((uint4*)l)[2] = ((const uint4*)g)[2];
    ((uint4*)l)[3] = ((const uint4*)g)[3];
#endif
  }
  __syncthreads();

  for (int s = 0; s < NSTAGES; ++s) {
    const char* cur = smem + (s & 1) * STAGE_BYTES;
    char* nxt = smem + ((s + 1) & 1) * STAGE_BYTES;
    const bool haveNext = (s + 1) < NSTAGES;

    // Issue next stage fill (overlaps with compute below). The target buffer
    // held stage s-1, whose readers finished before the previous barrier.
    const char* gn = Bbytes +
        (size_t)(colSplitBase + (s + 1) * STAGE_COLS + frow) * 256 + foff;
    char* ln = nxt + frow * STAGE_STRIDE + foff;
#if HAS_ASYNC_LDS
    if (haveNext) {
      __builtin_amdgcn_global_load_async_to_lds_b128((as1_v4i*)gn, (as3_v4i*)ln, 0, 0);
      __builtin_amdgcn_global_load_async_to_lds_b128((as1_v4i*)gn, (as3_v4i*)ln, 16, 0);
      __builtin_amdgcn_global_load_async_to_lds_b128((as1_v4i*)gn, (as3_v4i*)ln, 32, 0);
      __builtin_amdgcn_global_load_async_to_lds_b128((as1_v4i*)gn, (as3_v4i*)ln, 48, 0);
    }
#else
    uint4 t0, t1, t2, t3;
    if (haveNext) {
      t0 = ((const uint4*)gn)[0];
      t1 = ((const uint4*)gn)[1];
      t2 = ((const uint4*)gn)[2];
      t3 = ((const uint4*)gn)[3];
    }
#endif

    // Compute 4 x 16-column chunks from the current stage (LDS).
    #pragma unroll
    for (int cc = 0; cc < 4; ++cc) {
      const char* rb = cur + (cc * 16 + r) * STAGE_STRIDE + 16 * hi;
      v16h b[4];
      #pragma unroll
      for (int k = 0; k < 4; ++k) {
        v8h lo = *(const v8h*)(rb + 64 * k);
        v8h h8 = *(const v8h*)(rb + 64 * k + 32);
        b[k] = __builtin_shufflevector(lo, h8, 0, 1, 2, 3, 4, 5, 6, 7,
                                               8, 9, 10, 11, 12, 13, 14, 15);
      }

      v8f c = {0.f, 0.f, 0.f, 0.f, 0.f, 0.f, 0.f, 0.f};
      v8f d = {0.f, 0.f, 0.f, 0.f, 0.f, 0.f, 0.f, 0.f};
      #pragma unroll
      for (int k = 0; k < 4; ++k) {
        c = __builtin_amdgcn_wmma_f32_16x16x32_f16(false, a0[k], false, b[k],
                                                   (short)0, c, false, false);
        d = __builtin_amdgcn_wmma_f32_16x16x32_f16(false, a1[k], false, b[k],
                                                   (short)0, d, false, false);
      }

      // A pre-scaled by 2/ln2 -> contribution is exp2(acc). Batch TRANS ops.
      float e[16];
      #pragma unroll
      for (int v = 0; v < 8; ++v) {
        e[v]     = __builtin_amdgcn_exp2f(c[v]);
        e[8 + v] = __builtin_amdgcn_exp2f(d[v]);
      }
      #pragma unroll
      for (int v = 0; v < 8; ++v) {
        racc0[v] += e[v];
        racc1[v] += e[8 + v];
      }
    }

#if HAS_ASYNC_LDS
    wait_async0();          // our async fills for stage s+1 have landed
#else
    if (haveNext) {
      ((uint4*)ln)[0] = t0;
      ((uint4*)ln)[1] = t1;
      ((uint4*)ln)[2] = t2;
      ((uint4*)ln)[3] = t3;
    }
#endif
    __syncthreads();        // all fills visible; all reads of `cur` done
  }

  // Reduce across the 16 lanes of each half-wave (same hi group): xor masks
  // 1,2,4,8 never flip bit 4, so the reduction stays within the group.
  #pragma unroll
  for (int v = 0; v < 8; ++v) {
    #pragma unroll
    for (int m = 1; m < 16; m <<= 1) {
      racc0[v] += __shfl_xor(racc0[v], m, 32);
      racc1[v] += __shfl_xor(racc1[v], m, 32);
    }
  }
  if (r == 0) {
    #pragma unroll
    for (int v = 0; v < 8; ++v) {
      atomicAdd(&rowsum[rowA + 8 * hi + v], racc0[v]);
      atomicAdd(&rowsum[rowA + 16 + 8 * hi + v], racc1[v]);
    }
  }
}

// ---------------------------------------------------------------------------
// Kernel 3: loss = mean( log(denom + 1e-8) - dot/tau )
// ---------------------------------------------------------------------------
__global__ void __launch_bounds__(256)
loss_kernel(const float* __restrict__ rowsum, const float* __restrict__ posdot,
            float* __restrict__ out) {
  __shared__ float sh[256];
  float acc = 0.f;
  for (int i = threadIdx.x; i < NROWS; i += 256)
    acc += logf(rowsum[i] + 1e-8f) - posdot[i] * TAU_INV;
  sh[threadIdx.x] = acc;
  __syncthreads();
  #pragma unroll
  for (int s = 128; s > 0; s >>= 1) {
    if (threadIdx.x < s) sh[threadIdx.x] += sh[threadIdx.x + s];
    __syncthreads();
  }
  if (threadIdx.x == 0) out[0] = sh[0] / (float)NROWS;
}

// ---------------------------------------------------------------------------
extern "C" void kernel_launch(void* const* d_in, const int* in_sizes, int n_in,
                              void* d_out, int out_size, void* d_ws,
                              size_t ws_size, hipStream_t stream) {
  (void)in_sizes; (void)n_in; (void)out_size; (void)ws_size;
  const float* z1 = (const float*)d_in[0];
  const float* z2 = (const float*)d_in[1];
  float* out = (float*)d_out;

  // Workspace: A f16 (4MB) | B f16 (4MB) | posdot f32 | rowsum f32
  _Float16* Ah = (_Float16*)d_ws;
  _Float16* Bh = Ah + (size_t)NROWS * DIM;
  float* posdot = (float*)(Bh + (size_t)NROWS * DIM);
  float* rowsum = posdot + NROWS;

  norm_pack_kernel<<<NROWS / 8, 256, 0, stream>>>(z1, z2, Ah, Bh, posdot, rowsum);
  dim3 grid(NROWS / ROWS_PER_BLOCK, NSPLIT);
  simsum_kernel<<<grid, 256, 0, stream>>>(Ah, Bh, rowsum);
  loss_kernel<<<1, 256, 0, stream>>>(rowsum, posdot, out);
}
